// Encoder_51582557225716
// MI455X (gfx1250) — matrix operand order repaired and checked by reading, not compiled, not run on hardware
//
#include <hip/hip_runtime.h>
#include <math.h>
#include <stdint.h>

// ---------------- problem constants ----------------
constexpr int B  = 4;
constexpr int S  = 1024;
constexpr int D  = 1024;
constexpr int F  = 2816;
constexpr int L  = 4;
constexpr int H  = 16;
constexpr int DH = 64;
constexpr int M  = B * S;         // 4096 tokens

// ---------------- WMMA vector types ----------------
typedef __attribute__((ext_vector_type(16))) __bf16          v16bf;
typedef __attribute__((ext_vector_type(16))) unsigned short  v16us;
typedef __attribute__((ext_vector_type(8)))  float           v8f;

static __device__ __forceinline__ v16bf bc(v16us u) { return __builtin_bit_cast(v16bf, u); }

// ---------------- helpers ----------------
__device__ __forceinline__ unsigned short f2bf(float f) {
  unsigned int u = __float_as_uint(f);
  u += 0x7FFFu + ((u >> 16) & 1u);            // round-to-nearest-even
  return (unsigned short)(u >> 16);
}
__device__ __forceinline__ float bf2f(unsigned short h) {
  return __uint_as_float(((unsigned int)h) << 16);
}
__device__ __forceinline__ float gelu_tanh(float x) {
  float x3 = x * x * x;
  return 0.5f * x * (1.0f + tanhf(0.7978845608028654f * (x + 0.044715f * x3)));
}
// T5 relative position bias (NB=32, half=16, max_exact=8, MD=128); rel = k - q
__device__ __forceinline__ float t5_bias_rel(const float* __restrict__ rel_bias,
                                             int head, int rel) {
  int bucket = (rel > 0) ? 16 : 0;
  int arel = rel < 0 ? -rel : rel;
  if (arel < 8) {
    bucket += arel;
  } else {
    int large = 8 + (int)(logf((float)arel * 0.125f) * (8.0f / 2.772588722239781f)); // /log(16)
    bucket += (large > 15) ? 15 : large;
  }
  return rel_bias[bucket * H + head];
}

// ---------------- embedding lookup ----------------
__global__ void embed_kernel(const int* __restrict__ ids,
                             const float* __restrict__ emb,
                             float* __restrict__ h) {
  int t = blockIdx.x;
  int id = ids[t];
  const float* src = emb + (size_t)id * D;
  float* dst = h + (size_t)t * D;
  for (int i = threadIdx.x; i < D; i += 256) dst[i] = src[i];
}

// ---------------- f32 -> bf16 convert ----------------
__global__ void cvt_bf16_kernel(const float* __restrict__ in,
                                unsigned short* __restrict__ out, int n) {
  int i = blockIdx.x * 256 + threadIdx.x;
  if (i < n) out[i] = f2bf(in[i]);
}

// ---------------- RMSNorm ----------------
__global__ void rmsnorm_kernel(const float* __restrict__ x,
                               const float* __restrict__ w,
                               unsigned short* __restrict__ out_bf,
                               float* __restrict__ out_f, int useF32) {
  int t = blockIdx.x;
  const float* xp = x + (size_t)t * D;
  __shared__ float red[256];
  float s = 0.f;
  for (int i = threadIdx.x; i < D; i += 256) { float v = xp[i]; s += v * v; }
  red[threadIdx.x] = s;
  __syncthreads();
  for (int o = 128; o > 0; o >>= 1) {
    if (threadIdx.x < o) red[threadIdx.x] += red[threadIdx.x + o];
    __syncthreads();
  }
  float scale = rsqrtf(red[0] * (1.0f / D) + 1e-6f);
  for (int i = threadIdx.x; i < D; i += 256) {
    float v = xp[i] * scale * w[i];
    if (useF32) out_f[(size_t)t * D + i] = v;
    else        out_bf[(size_t)t * D + i] = f2bf(v);
  }
}

// ---------------- fused epilogue (compile-time selected) ----------------
// 0 = store bf16, 1 = gelu->bf16, 2 = aux(bf16)*acc->bf16, 3 = aux(f32)+acc->f32
template <int EPI>
__device__ __forceinline__ void epi_store(void* __restrict__ out,
                                          const void* __restrict__ aux,
                                          size_t idx, float v) {
  if constexpr (EPI == 0) {
    ((unsigned short*)out)[idx] = f2bf(v);
  } else if constexpr (EPI == 1) {
    ((unsigned short*)out)[idx] = f2bf(gelu_tanh(v));
  } else if constexpr (EPI == 2) {
    ((unsigned short*)out)[idx] = f2bf(bf2f(((const unsigned short*)aux)[idx]) * v);
  } else {
    ((float*)out)[idx] = ((const float*)aux)[idx] + v;
  }
}

// ---------------- tiled WMMA GEMM: C[M,N] = A[M,K](bf16) x W[N,K](f32)^T ----
// block tile 64x128, 8 waves, wave tile 32x32 (4 wmma / K-step).
// Double-buffered LDS: tile k+1 is staged (async global->LDS A copy + W loads)
// while tile k is consumed by WMMA; one barrier per K-step.
template <int EPI, int NDIM, int KDIM>
__global__ __launch_bounds__(256)
void gemm_kernel(const unsigned short* __restrict__ A,
                 const float* __restrict__ W,
                 void* __restrict__ out, const void* __restrict__ aux) {
  __shared__ unsigned short As[2][64][40];    // 80B row stride (16B aligned)
  __shared__ unsigned short Bs[2][128][40];

  const int m0 = blockIdx.y * 64;
  const int n0 = blockIdx.x * 128;
  const int tid  = threadIdx.x;
  const int lane = tid & 31;
  const int w    = tid >> 5;               // 8 waves (wave32)
  const int mW = (w & 1) * 32;             // wave M offset within block tile
  const int nW = (w >> 1) * 32;            // wave N offset within block tile

  v8f acc00 = {}, acc01 = {}, acc10 = {}, acc11 = {};

  const int arow = tid >> 2, acol = (tid & 3) * 8;   // A stage: 64 x 32 bf16
  const int brow = tid >> 1, bcol = (tid & 1) * 16;  // W stage: 128 x 32 f32->bf16

  const unsigned short* aRowPtr = A + (size_t)(m0 + arow) * KDIM + acol;
  const float*          wRowPtr = W + (size_t)(n0 + brow) * KDIM + bcol;
  const unsigned aLds0 = (unsigned)(uintptr_t)&As[0][arow][acol];  // 16B-aligned
  const unsigned aLds1 = (unsigned)(uintptr_t)&As[1][arow][acol];

  // ---- prologue: stage tile 0 into buffer 0 ----
  asm volatile("global_load_async_to_lds_b128 %0, %1, off"
               :: "v"(aLds0), "v"(aRowPtr) : "memory");
  {
    const float4* wp = (const float4*)wRowPtr;
    float4 f0 = wp[0], f1 = wp[1], f2 = wp[2], f3 = wp[3];
    unsigned short* bp = &Bs[0][brow][bcol];
    bp[0]  = f2bf(f0.x); bp[1]  = f2bf(f0.y); bp[2]  = f2bf(f0.z); bp[3]  = f2bf(f0.w);
    bp[4]  = f2bf(f1.x); bp[5]  = f2bf(f1.y); bp[6]  = f2bf(f1.z); bp[7]  = f2bf(f1.w);
    bp[8]  = f2bf(f2.x); bp[9]  = f2bf(f2.y); bp[10] = f2bf(f2.z); bp[11] = f2bf(f2.w);
    bp[12] = f2bf(f3.x); bp[13] = f2bf(f3.y); bp[14] = f2bf(f3.z); bp[15] = f2bf(f3.w);
  }
  asm volatile("s_wait_asynccnt 0" ::: "memory");
  __syncthreads();

  int ping = 0;
#pragma unroll 1
  for (int k0 = 0; k0 < KDIM; k0 += 32) {
    const bool hasNext = (k0 + 32) < KDIM;
    float4 f0 = {}, f1 = {}, f2 = {}, f3 = {};
    if (hasNext) {
      // issue next tile's async A copy + W global loads before computing
      asm volatile("global_load_async_to_lds_b128 %0, %1, off"
                   :: "v"(ping ? aLds0 : aLds1), "v"(aRowPtr + k0 + 32) : "memory");
      const float4* wp = (const float4*)(wRowPtr + k0 + 32);
      f0 = wp[0]; f1 = wp[1]; f2 = wp[2]; f3 = wp[3];
      __builtin_prefetch(wRowPtr + k0 + 64, 0, 0);   // two tiles ahead
    }

    // ---- compute on buf[ping] ----
    v16us a0u, a1u, b0u, b1u;
    const int rA0 = mW + (lane & 15), rA1 = rA0 + 16;
    const int kbA = (lane >> 4) * 8;
#pragma unroll
    for (int j = 0; j < 8; ++j) {
      a0u[j] = As[ping][rA0][kbA + j];  a0u[8 + j] = As[ping][rA0][kbA + 16 + j];
      a1u[j] = As[ping][rA1][kbA + j];  a1u[8 + j] = As[ping][rA1][kbA + 16 + j];
    }
    const int rB0 = nW + (lane & 15), rB1 = rB0 + 16;
    const int kbB = (lane >> 4) * 8;
#pragma unroll
    for (int j = 0; j < 8; ++j) {
      b0u[j] = Bs[ping][rB0][kbB + j];  b0u[8 + j] = Bs[ping][rB0][kbB + 16 + j];
      b1u[j] = Bs[ping][rB1][kbB + j];  b1u[8 + j] = Bs[ping][rB1][kbB + 16 + j];
    }
    v16bf a0 = bc(a0u), a1 = bc(a1u), b0 = bc(b0u), b1 = bc(b1u);
    acc00 = __builtin_amdgcn_wmma_f32_16x16x32_bf16(false, a0, false, b0, (short)0, acc00, false, false);
    acc01 = __builtin_amdgcn_wmma_f32_16x16x32_bf16(false, a0, false, b1, (short)0, acc01, false, false);
    acc10 = __builtin_amdgcn_wmma_f32_16x16x32_bf16(false, a1, false, b0, (short)0, acc10, false, false);
    acc11 = __builtin_amdgcn_wmma_f32_16x16x32_bf16(false, a1, false, b1, (short)0, acc11, false, false);

    // ---- finish staging next tile (convert W regs -> LDS) ----
    if (hasNext) {
      unsigned short* bp = &Bs[ping ^ 1][brow][bcol];
      bp[0]  = f2bf(f0.x); bp[1]  = f2bf(f0.y); bp[2]  = f2bf(f0.z); bp[3]  = f2bf(f0.w);
      bp[4]  = f2bf(f1.x); bp[5]  = f2bf(f1.y); bp[6]  = f2bf(f1.z); bp[7]  = f2bf(f1.w);
      bp[8]  = f2bf(f2.x); bp[9]  = f2bf(f2.y); bp[10] = f2bf(f2.z); bp[11] = f2bf(f2.w);
      bp[12] = f2bf(f3.x); bp[13] = f2bf(f3.y); bp[14] = f2bf(f3.z); bp[15] = f2bf(f3.w);
    }
    asm volatile("s_wait_asynccnt 0" ::: "memory");
    __syncthreads();                       // single barrier per K-step
    ping ^= 1;
  }

  // epilogue: C/D layout -> lane half selects M+8, N = lane&15
  const int mB = m0 + mW + ((lane >> 4) ? 8 : 0);
  const int nB = n0 + nW + (lane & 15);
#pragma unroll
  for (int r = 0; r < 8; ++r) {
    epi_store<EPI>(out, aux, (size_t)(mB + r)      * NDIM + nB,      acc00[r]);
    epi_store<EPI>(out, aux, (size_t)(mB + r)      * NDIM + nB + 16, acc01[r]);
    epi_store<EPI>(out, aux, (size_t)(mB + 16 + r) * NDIM + nB,      acc10[r]);
    epi_store<EPI>(out, aux, (size_t)(mB + 16 + r) * NDIM + nB + 16, acc11[r]);
  }
}

// ---------------- attention: 32 query rows per block, full-row f32 softmax in LDS
// LDS: sc[32][1024] f32 | red[256] | rmax[32] | rsum[32] | Vs[64][40] bf16 | biasTab[2048] f32
template <bool HASBIAS>
__global__ __launch_bounds__(256)
void attn_kernel(const unsigned short* __restrict__ Q,
                 const unsigned short* __restrict__ Kb,
                 const unsigned short* __restrict__ Vb,
                 unsigned short* __restrict__ ctx,
                 const float* __restrict__ rel_bias) {
  extern __shared__ float smem[];
  float* sc   = smem;                               // [32][1024] scores, 128 KB
  float* red  = smem + 32 * S;                      // [256]
  float* rmax = red + 256;                          // [32]
  float* rsum = rmax + 32;                          // [32]
  unsigned short* Vs = (unsigned short*)(rsum + 32);// [64][40] transposed V tile
  float* biasTab = (float*)(Vs + 64 * 40);          // [2048] bias vs (k - q + 1023)

  const int b    = blockIdx.z;
  const int head = blockIdx.y;
  const int q0   = blockIdx.x * 32;
  const int tid  = threadIdx.x;
  const int lane = tid & 31;
  const int w    = tid >> 5;

  // ---- phase 0: build per-head bias table (one logf sweep per block) ----
  if constexpr (HASBIAS) {
    for (int idx = tid; idx < 2 * S - 1; idx += 256)
      biasTab[idx] = t5_bias_rel(rel_bias, head, idx - (S - 1));
  }
  __syncthreads();

  // ---- phase 1: scores = Q K^T (+bias) ----
  {
    // preload the 4 Q fragments (2 m-tiles x 2 K-steps) into registers
    v16us aq[2][2];
#pragma unroll
    for (int mt = 0; mt < 2; ++mt) {
      const unsigned short* qp =
          Q + ((size_t)(b * S + q0 + mt * 16 + (lane & 15))) * D + head * DH;
#pragma unroll
      for (int ksi = 0; ksi < 2; ++ksi) {
        const int kbA = ksi * 32 + (lane >> 4) * 8;
        union { uint4 q[2]; v16us v; } u;
        u.q[0] = *(const uint4*)(qp + kbA);
        u.q[1] = *(const uint4*)(qp + kbA + 16);
        aq[mt][ksi] = u.v;
      }
    }
    for (int nt = w; nt < 64; nt += 8) {
      const int kn = nt * 16 + (lane & 15);
      const unsigned short* kp = Kb + ((size_t)(b * S + kn)) * D + head * DH;
      v8f acc0 = {}, acc1 = {};
#pragma unroll
      for (int ksi = 0; ksi < 2; ++ksi) {
        const int kbB = ksi * 32 + (lane >> 4) * 8;
        union { uint4 q[2]; v16us v; } ub;
        ub.q[0] = *(const uint4*)(kp + kbB);
        ub.q[1] = *(const uint4*)(kp + kbB + 16);
        v16bf bb = bc(ub.v);
        acc0 = __builtin_amdgcn_wmma_f32_16x16x32_bf16(false, bc(aq[0][ksi]), false, bb, (short)0, acc0, false, false);
        acc1 = __builtin_amdgcn_wmma_f32_16x16x32_bf16(false, bc(aq[1][ksi]), false, bb, (short)0, acc1, false, false);
      }
      const int n  = nt * 16 + (lane & 15);
      const int mB = (lane >> 4) ? 8 : 0;
#pragma unroll
      for (int r = 0; r < 8; ++r) {
        float v0 = acc0[r], v1 = acc1[r];
        if constexpr (HASBIAS) {
          v0 += biasTab[n - (q0 + mB + r) + (S - 1)];
          v1 += biasTab[n - (q0 + 16 + mB + r) + (S - 1)];
        }
        sc[(mB + r) * S + n]      = v0;
        sc[(16 + mB + r) * S + n] = v1;
      }
    }
  }
  __syncthreads();

  // ---- phase 2: f32 softmax over each row of 1024 (float4 sweeps) ----
  {
    const int row = tid >> 3, sub = tid & 7;
    float4* base4 = (float4*)(sc + row * S + sub * 128);   // 32 float4s
    float lmax = -1e30f;
    for (int i = 0; i < 32; ++i) {
      float4 v = base4[i];
      lmax = fmaxf(lmax, fmaxf(fmaxf(v.x, v.y), fmaxf(v.z, v.w)));
    }
    red[tid] = lmax;
    __syncthreads();
    if (sub == 0) {
      float mx = red[tid];
      for (int j = 1; j < 8; ++j) mx = fmaxf(mx, red[tid + j]);
      rmax[row] = mx;
    }
    __syncthreads();
    const float rm = rmax[row];
    float ls = 0.f;
    for (int i = 0; i < 32; ++i) {
      float4 v = base4[i];
      v.x = __expf(v.x - rm); v.y = __expf(v.y - rm);
      v.z = __expf(v.z - rm); v.w = __expf(v.w - rm);
      base4[i] = v;
      ls += v.x + v.y + v.z + v.w;
    }
    red[tid] = ls;
    __syncthreads();
    if (sub == 0) {
      float s2 = 0.f;
      for (int j = 0; j < 8; ++j) s2 += red[tid + j];
      rsum[row] = s2;
    }
    __syncthreads();
    const float inv = 1.0f / rsum[row];
    for (int i = 0; i < 32; ++i) {
      float4 v = base4[i];
      v.x *= inv; v.y *= inv; v.z *= inv; v.w *= inv;
      base4[i] = v;
    }
  }
  __syncthreads();

  // ---- phase 3: ctx = P V  (K = 1024), V tile staged transposed in LDS ----
  {
    const int mt = w & 1, nt3 = w >> 1;      // 2 x 4 output tiles of 16x16
    const int rA  = mt * 16 + (lane & 15);
    const int dhc = nt3 * 16 + (lane & 15);
    const int keyL = tid >> 3;               // 0..31 staged key row
    const int dh0  = (tid & 7) * 8;          // 0..56 staged dh columns
    v8f acc = {};
#pragma unroll 1
    for (int ks = 0; ks < S; ks += 32) {
      // cooperative stage: V[ks+keyL][dh0..dh0+7] -> Vs[dh][keyL] (transposed)
      union { uint4 q; unsigned short u[8]; } vv;
      vv.q = *(const uint4*)(Vb + ((size_t)(b * S + ks + keyL)) * D + head * DH + dh0);
#pragma unroll
      for (int j = 0; j < 8; ++j) Vs[(dh0 + j) * 40 + keyL] = vv.u[j];
      __syncthreads();

      v16us au, bu;
      const int kbA = ks + (lane >> 4) * 8;
#pragma unroll
      for (int j = 0; j < 8; ++j) {
        au[j]     = f2bf(sc[rA * S + kbA + j]);
        au[8 + j] = f2bf(sc[rA * S + kbA + 16 + j]);
      }
      const int kbB = (lane >> 4) * 8;       // local key base within tile
#pragma unroll
      for (int j = 0; j < 8; ++j) { bu[j] = Vs[dhc * 40 + kbB + j]; bu[8 + j] = Vs[dhc * 40 + kbB + 16 + j]; }
      acc = __builtin_amdgcn_wmma_f32_16x16x32_bf16(
          false, bc(au), false, bc(bu), (short)0, acc, false, false);
      __syncthreads();
    }
    const int mBase = mt * 16 + ((lane >> 4) ? 8 : 0);
#pragma unroll
    for (int r = 0; r < 8; ++r) {
      int mq = q0 + mBase + r;
      ctx[((size_t)(b * S + mq)) * D + head * DH + dhc] = f2bf(acc[r]);
    }
  }
}

// ---------------- host orchestration ----------------
extern "C" void kernel_launch(void* const* d_in, const int* in_sizes, int n_in,
                              void* d_out, int out_size, void* d_ws, size_t ws_size,
                              hipStream_t stream) {
  (void)in_sizes; (void)n_in; (void)out_size; (void)ws_size;
  const int*   ids      = (const int*)d_in[0];
  const float* memory   = (const float*)d_in[1];
  const float* embed    = (const float*)d_in[2];
  const float* rel_bias = (const float*)d_in[3];
  const float* q_s = (const float*)d_in[4];
  const float* k_s = (const float*)d_in[5];
  const float* v_s = (const float*)d_in[6];
  const float* o_s = (const float*)d_in[7];
  const float* ln1 = (const float*)d_in[8];
  const float* q_c = (const float*)d_in[9];
  const float* k_c = (const float*)d_in[10];
  const float* v_c = (const float*)d_in[11];
  const float* o_c = (const float*)d_in[12];
  const float* ln2 = (const float*)d_in[13];
  const float* wi0 = (const float*)d_in[14];
  const float* wi1 = (const float*)d_in[15];
  const float* wo  = (const float*)d_in[16];
  const float* ln3 = (const float*)d_in[17];
  const float* ln_f = (const float*)d_in[18];

  // workspace carve-up
  char* p = (char*)d_ws;
  float* h = (float*)p;                      p += (size_t)M * D * sizeof(float);
  unsigned short* xn   = (unsigned short*)p; p += (size_t)M * D * 2;
  unsigned short* qb   = (unsigned short*)p; p += (size_t)M * D * 2;
  unsigned short* kb   = (unsigned short*)p; p += (size_t)M * D * 2;
  unsigned short* vb   = (unsigned short*)p; p += (size_t)M * D * 2;
  unsigned short* ctx  = (unsigned short*)p; p += (size_t)M * D * 2;
  unsigned short* memb = (unsigned short*)p; p += (size_t)M * D * 2;
  unsigned short* ffn  = (unsigned short*)p; p += (size_t)M * F * 2;

  const dim3 blk(256);
  const dim3 gD(D / 128, M / 64);       // GEMMs with N = D  -> (8, 64)
  const dim3 gF(F / 128, M / 64);       // GEMMs with N = F  -> (22, 64)
  const dim3 gA(S / 32, H, B);          // attention blocks
  const size_t attn_shmem =
      (size_t)(32 * S + 256 + 32 + 32) * sizeof(float)   // scores + reductions
      + (size_t)64 * 40 * 2                              // V tile
      + (size_t)2048 * sizeof(float);                    // bias table

  // h = embed[ids]; memb = bf16(memory)
  embed_kernel<<<M, blk, 0, stream>>>(ids, embed, h);
  cvt_bf16_kernel<<<(M * D + 255) / 256, blk, 0, stream>>>(memory, memb, M * D);

  for (int l = 0; l < L; ++l) {
    const float* Wq  = q_s + (size_t)l * D * D;
    const float* Wk  = k_s + (size_t)l * D * D;
    const float* Wv  = v_s + (size_t)l * D * D;
    const float* Wo  = o_s + (size_t)l * D * D;
    const float* Wqc = q_c + (size_t)l * D * D;
    const float* Wkc = k_c + (size_t)l * D * D;
    const float* Wvc = v_c + (size_t)l * D * D;
    const float* Woc = o_c + (size_t)l * D * D;
    const float* Wi0 = wi0 + (size_t)l * F * D;
    const float* Wi1 = wi1 + (size_t)l * F * D;
    const float* Wff = wo  + (size_t)l * D * F;

    // ---- self attention ----
    rmsnorm_kernel<<<M, blk, 0, stream>>>(h, ln1 + (size_t)l * D, xn, nullptr, 0);
    gemm_kernel<0, D, D><<<gD, blk, 0, stream>>>(xn, Wq, qb, nullptr);
    gemm_kernel<0, D, D><<<gD, blk, 0, stream>>>(xn, Wk, kb, nullptr);
    gemm_kernel<0, D, D><<<gD, blk, 0, stream>>>(xn, Wv, vb, nullptr);
    attn_kernel<true><<<gA, blk, attn_shmem, stream>>>(qb, kb, vb, ctx, rel_bias);
    gemm_kernel<3, D, D><<<gD, blk, 0, stream>>>(ctx, Wo, h, h);   // residual

    // ---- cross attention ----
    rmsnorm_kernel<<<M, blk, 0, stream>>>(h, ln2 + (size_t)l * D, xn, nullptr, 0);
    gemm_kernel<0, D, D><<<gD, blk, 0, stream>>>(xn,   Wqc, qb, nullptr);
    gemm_kernel<0, D, D><<<gD, blk, 0, stream>>>(memb, Wkc, kb, nullptr);
    gemm_kernel<0, D, D><<<gD, blk, 0, stream>>>(memb, Wvc, vb, nullptr);
    attn_kernel<false><<<gA, blk, attn_shmem, stream>>>(qb, kb, vb, ctx, rel_bias);
    gemm_kernel<3, D, D><<<gD, blk, 0, stream>>>(ctx, Woc, h, h);  // residual

    // ---- gated-GELU FFN ----
    rmsnorm_kernel<<<M, blk, 0, stream>>>(h, ln3 + (size_t)l * D, xn, nullptr, 0);
    gemm_kernel<1, F, D><<<gF, blk, 0, stream>>>(xn, Wi0, ffn, nullptr);  // gelu
    gemm_kernel<2, F, D><<<gF, blk, 0, stream>>>(xn, Wi1, ffn, ffn);      // gate mul
    gemm_kernel<3, D, F><<<gD, blk, 0, stream>>>(ffn, Wff, h, h);         // residual
  }

  // final norm -> f32 output
  rmsnorm_kernel<<<M, blk, 0, stream>>>(h, ln_f, nullptr, (float*)d_out, 1);
}
